// AdaptiveWaveletBank_56573309224236
// MI455X (gfx1250) — matrix operand order repaired and checked by reading, not compiled, not run on hardware
//
#include <hip/hip_runtime.h>

// ---------------------------------------------------------------------------
// Wavelet bank as Toeplitz GEMM on CDNA5 WMMA (gfx1250, wave32), with
// TDM (tensor_load_to_lds) / async-to-LDS staging of the signal window.
//
//   out[b, s, n] = sum_k signal[b, n - wl + k] * w_s[k]        (re and im)
//
// Per block: one scale s, all 16 batches (== WMMA M), TN=256 positions
// (16 column tiles, two per wave, 8 waves).  Contraction over absolute
// signal position m in K-chunks of 32:
//   A[b][kk]  = (f16) sigf32[b][m0 + kk]       (LDS f32, cvt at frag build)
//   B[kk][j]  = taps[32c + kk - j]             (Toeplitz, tile-independent)
//   D         = v_wmma_f32_16x16x32_f16(A, B, C)   f32 accumulation
//
// Signal staging:
//   interior blocks -> ONE TENSOR_LOAD_TO_LDS (iterate mode: 16 x 1-row
//                      tiles, global += L, lds += srow)  [TENSORcnt]
//   boundary blocks -> per-element global_load_async_to_lds_b32 for the
//                      in-range span + scalar zero-fill    [ASYNCcnt]
// ---------------------------------------------------------------------------

typedef __attribute__((ext_vector_type(16))) _Float16     v16h;
typedef __attribute__((ext_vector_type(8)))  float        v8f;
typedef __attribute__((ext_vector_type(4)))  unsigned int v4u;
typedef __attribute__((ext_vector_type(8)))  int          v8i;
typedef __attribute__((ext_vector_type(4)))  int          v4i;

#define NUM_SCALES 16
#define TN         256
#define WLMAX      2048
#define WMAX       (WLMAX + TN + 32)            // 2336 f32 window max
#define SROWF_MAX  (((WMAX + 63) & ~63) + 4)    // 2372: mult of 4, ==4 mod 64
#define TAPS_PADF  16
#define TAPS_LEN   (TAPS_PADF + WLMAX + 64)     // 2128

__device__ __forceinline__ void cvt4(_Float16* d, float4 v) {
    d[0] = (_Float16)v.x; d[1] = (_Float16)v.y;
    d[2] = (_Float16)v.z; d[3] = (_Float16)v.w;
}

__global__ __launch_bounds__(256)
void AdaptiveWaveletBank_wmma_kernel(const float* __restrict__ signal,
                                     const float* __restrict__ scales_log,
                                     const float* __restrict__ cf_log,
                                     const float* __restrict__ bw_log,
                                     float* __restrict__ out, int L)
{
    __shared__ __align__(16) float    sigf[16 * SROWF_MAX];   // ~148 KB
    __shared__ __align__(16) _Float16 tapsR[TAPS_LEN];        // ~4.3 KB
    __shared__ __align__(16) _Float16 tapsI[TAPS_LEN];        // ~4.3 KB

    const int s   = blockIdx.y;
    const int n0  = blockIdx.x * TN;
    const int tid = threadIdx.x;

    // ---- per-scale wavelet parameters ----
    const float sc = __expf(scales_log[s]);
    const float cf = __expf(cf_log[0]);
    const float bw = __expf(bw_log[0]);
    int wl = (int)(64.0f * sc);
    wl = (wl < L / 2) ? wl : (L / 2);
    wl = (wl > 8) ? wl : 8;
    wl += (wl & 1);

    const int base = n0 - wl;
    const int W    = wl + TN + 32;                 // window width (f32 elems)
    const int Wp4  = (W + 3) & ~3;                 // 16B-granular copy width
    const int srow = ((W + 63) & ~63) + 4;         // LDS row pitch (f32 elems)

    // ---- taps in LDS (zero-padded for Toeplitz reads); norm==1+1e-8 ----
    const float inv_t    = 1.0f / (bw * fmaxf(sc, 0.1f));
    const float w0       = (2.0f * 3.14159265358979f / 6.0f) * cf;
    const float inv_norm = 1.0f / (1.0f + 1e-8f);
    for (int k = tid; k < TAPS_LEN; k += 256) {
        float wr = 0.0f, wi = 0.0f;
        const int kk = k - TAPS_PADF;
        if (kk >= 0 && kk < wl) {
            const float t   = (float)kk * inv_t;
            const float env = __expf(-0.5f * t * t) * inv_norm;
            const float ph  = w0 * t;
            wr = env * __cosf(ph);
            wi = env * __sinf(ph);
        }
        tapsR[k] = (_Float16)wr;
        tapsI[k] = (_Float16)wi;
    }

    // ---- stage signal window: TDM for interior blocks, async for edges ----
    const bool interior = (base >= 0) && (base + Wp4 <= L);
#if defined(__gfx1250__)
    if (interior) {
        if (tid < 32) {   // one wave issues the DMA; EXEC is ignored by TDM
            const unsigned long long ga =
                (unsigned long long)(const void*)(signal + base);
            const unsigned int lds = (unsigned int)(unsigned long long)(void*)sigf;
            // D# group0: count=1 | lds_addr | global_addr | type=2
            v4u g0 = { 1u, lds, (unsigned int)ga,
                       (unsigned int)((ga >> 32) & 0x01FFFFFFu) | (2u << 30) };
            // D# group1: data_size=4B, iterate_enable, tensor_dim0=huge,
            //            tensor_dim1=1, tile_dim0=Wp4 (1-row tile)
            const unsigned int td0 = 0x7FFFFFFFu;
            v8i g1 = { (int)((2u << 16) | (1u << 19)),
                       (int)((td0 & 0xFFFFu) << 16),
                       (int)(((td0 >> 16) & 0xFFFFu) | (1u << 16)),
                       (int)((unsigned int)Wp4 << 16),
                       0,
                       (int)(unsigned int)L,        // tensor_dim0_stride
                       0, 0 };
            // D# group2 (iterate mode): lds_addr_inc=srow, glb_addr_inc=L,
            //            iterate_count=15 -> 16 rows
            v4i g2 = { 1, (int)(unsigned int)srow, (int)(unsigned int)L,
                       (int)(15u << 16) };
            v4i g3 = { 0, 0, 0, 0 };
#if __has_builtin(__builtin_amdgcn_tensor_load_to_lds)
#if __has_include(<hip/amd_detail/amd_gfx1250_TDM.h>)
            v8i g4 = { 0, 0, 0, 0, 0, 0, 0, 0 };
            __builtin_amdgcn_tensor_load_to_lds(g0, g1, g2, g3, g4, 0);
#else
            __builtin_amdgcn_tensor_load_to_lds(g0, g1, g2, g3, 0);
#endif
            __builtin_amdgcn_s_wait_tensorcnt(0);
#endif
        }
    } else {
        // boundary blocks: async per-element copy of in-range span, zero-fill rest
        for (int r = 0; r < 16; ++r) {
            const float* srcrow = signal + (size_t)r * (size_t)L;
            float*       dstrow = sigf + r * srow;
            for (int i = tid; i < W; i += 256) {
                const int m = base + i;
                if (m >= 0 && m < L) {
                    const unsigned long long ga =
                        (unsigned long long)(const void*)(srcrow + m);
                    const unsigned int lds =
                        (unsigned int)(unsigned long long)(void*)(dstrow + i);
                    asm volatile("global_load_async_to_lds_b32 %0, %1, off"
                                 :: "v"(lds), "v"(ga) : "memory");
                } else {
                    dstrow[i] = 0.0f;
                }
            }
        }
        asm volatile("s_wait_asynccnt 0" ::: "memory");
    }
#else
    // host-pass / non-gfx1250 fallback: plain staging
    for (int r = 0; r < 16; ++r) {
        const float* srcrow = signal + (size_t)r * (size_t)L;
        float*       dstrow = sigf + r * srow;
        for (int i = tid; i < W; i += 256) {
            const int m = base + i;
            dstrow[i] = (m >= 0 && m < L) ? srcrow[m] : 0.0f;
        }
    }
#endif
    __syncthreads();

    // ---- WMMA main loop: 8 waves x 2 column-tiles each ----
    const int wave = tid >> 5;
    const int lane = tid & 31;
    const int lh   = lane >> 4;
    const int ln   = lane & 15;
    const int t0   = wave * 2;
    const int t1   = t0 + 1;

    v8f accR0 = {}, accI0 = {}, accR1 = {}, accI1 = {};
    const int chunks = (wl + 16 + 31) >> 5;

    for (int c = 0; c < chunks; ++c) {
        // B fragment (32x16 f16): Toeplitz taps, B[kk][j] = w[32c + kk - j]
        union { v16h v; _Float16 h[16]; } bR, bI;
        const int idx0 = TAPS_PADF + 32 * c + lh * 8 - ln;
        #pragma unroll
        for (int i = 0; i < 16; ++i) {
            bR.h[i] = tapsR[idx0 + i + (i & 8)];
            bI.h[i] = tapsI[idx0 + i + (i & 8)];
        }

        // A fragments: f32 LDS b128 reads + cvt to f16 (co-executes with XDL)
        {
            union { v16h v; _Float16 h[16]; } a;
            const float4* rp = (const float4*)(sigf + ln * srow + 16 * t0 + 32 * c);
            cvt4(&a.h[0],  rp[2 * lh]);
            cvt4(&a.h[4],  rp[2 * lh + 1]);
            cvt4(&a.h[8],  rp[4 + 2 * lh]);
            cvt4(&a.h[12], rp[5 + 2 * lh]);
            accR0 = __builtin_amdgcn_wmma_f32_16x16x32_f16(false, a.v, false, bR.v,
                                                           (short)0, accR0, false, false);
            accI0 = __builtin_amdgcn_wmma_f32_16x16x32_f16(false, a.v, false, bI.v,
                                                           (short)0, accI0, false, false);
        }
        {
            union { v16h v; _Float16 h[16]; } a;
            const float4* rp = (const float4*)(sigf + ln * srow + 16 * t1 + 32 * c);
            cvt4(&a.h[0],  rp[2 * lh]);
            cvt4(&a.h[4],  rp[2 * lh + 1]);
            cvt4(&a.h[8],  rp[4 + 2 * lh]);
            cvt4(&a.h[12], rp[5 + 2 * lh]);
            accR1 = __builtin_amdgcn_wmma_f32_16x16x32_f16(false, a.v, false, bR.v,
                                                           (short)0, accR1, false, false);
            accI1 = __builtin_amdgcn_wmma_f32_16x16x32_f16(false, a.v, false, bI.v,
                                                           (short)0, accI1, false, false);
        }
    }

    // ---- epilogue: C/D layout -> interleaved complex64 b64 stores ----
    const size_t Ls = (size_t)L;
    #pragma unroll
    for (int i = 0; i < 8; ++i) {
        const int b = i + lh * 8;
        const size_t rowbase = ((size_t)b * NUM_SCALES + (size_t)s) * Ls;
        {
            const size_t pos = (size_t)(n0 + 16 * t0 + ln);
            float2 v2 = make_float2(accR0[i], accI0[i]);
            *(float2*)(out + (rowbase + pos) * 2) = v2;
        }
        {
            const size_t pos = (size_t)(n0 + 16 * t1 + ln);
            float2 v2 = make_float2(accR1[i], accI1[i]);
            *(float2*)(out + (rowbase + pos) * 2) = v2;
        }
    }
}

extern "C" void kernel_launch(void* const* d_in, const int* in_sizes, int n_in,
                              void* d_out, int out_size, void* d_ws, size_t ws_size,
                              hipStream_t stream) {
    const float* signal     = (const float*)d_in[0];
    const float* scales_log = (const float*)d_in[1];
    const float* cf_log     = (const float*)d_in[2];
    const float* bw_log     = (const float*)d_in[3];
    float* outp = (float*)d_out;

    const int B = 16;
    const int L = in_sizes[0] / B;               // 32768

    dim3 grid(L / TN, NUM_SCALES);               // (128, 16)
    AdaptiveWaveletBank_wmma_kernel<<<grid, 256, 0, stream>>>(
        signal, scales_log, cf_log, bw_log, outp, L);
}
